// GINModel_67095979099186
// MI455X (gfx1250) — matrix operand order repaired and checked by reading, not compiled, not run on hardware
//
#include <hip/hip_runtime.h>
#include <hip/hip_bf16.h>
#include <stdint.h>

#define N_NODES 50000
#define N_EDGES 800000
#define N_GRAPHS 512
#define DIM 128
#define NLAYERS 3
#define BN_EPS_F 1e-5f

#define BROWS 32
#define LDS_STRIDE 136   // ushorts: 128 + 8 pad; 272B row stride (16B-aligned)

typedef __attribute__((ext_vector_type(16))) __bf16 v16bf;
typedef __attribute__((ext_vector_type(8)))  float  v8f;

union BF16Frag { v16bf v; uint4 q[2]; };

__device__ __forceinline__ unsigned short f2bf(float f) {
  unsigned u = __float_as_uint(f);
  u += 0x7fffu + ((u >> 16) & 1u);   // round-to-nearest-even
  return (unsigned short)(u >> 16);
}

// A fragment (16x32 bf16) from row-major [M][K] tile, row base = &tile[m*stride]
__device__ __forceinline__ v16bf load_a_frag(const unsigned short* rowp, int kbase, int lhalf) {
  BF16Frag f;
  f.q[0] = *(const uint4*)(rowp + kbase + (lhalf << 3));        // e0..7
  f.q[1] = *(const uint4*)(rowp + kbase + 16 + (lhalf << 3));   // e8..15
  return f.v;
}

// B fragment (32x16 bf16) from N-major weight row (&Wt[n*DIM]), K contiguous
__device__ __forceinline__ v16bf load_b_frag(const unsigned short* wrow, int kbase, int lhalf) {
  BF16Frag f;
  const unsigned short* p = wrow + kbase + (lhalf << 3);
  f.q[0] = *(const uint4*)(p);
  f.q[1] = *(const uint4*)(p + 16);
  return f.v;
}

// ------------------- one-time per launch: W (l,k,n) fp32 -> Wt (l,n,k) bf16 ----
__global__ void wconv_kernel(const float* __restrict__ W1, const float* __restrict__ W2,
                             unsigned short* __restrict__ Wt1, unsigned short* __restrict__ Wt2) {
  int i = blockIdx.x * blockDim.x + threadIdx.x;
  if (i >= NLAYERS * DIM * DIM) return;
  int l = i >> 14, rem = i & 16383;
  int k = rem >> 7, n = rem & 127;
  int dst = (l << 14) + (n << 7) + k;
  Wt1[dst] = f2bf(W1[i]);
  Wt2[dst] = f2bf(W2[i]);
}

// ---------------------------------------------------------------- z = (1+eps)*h
__global__ void zinit_kernel(const float* __restrict__ hin, float* __restrict__ z,
                             const float* __restrict__ eps, int layer) {
  int i = blockIdx.x * blockDim.x + threadIdx.x;   // over N*D/4
  if (i >= N_NODES * DIM / 4) return;
  float s = 1.0f + eps[layer];
  float4 v = ((const float4*)hin)[i];
  v.x *= s; v.y *= s; v.z *= s; v.w *= s;
  ((float4*)z)[i] = v;
}

// ------------------------------------------------- z[dst] += h[src] (wave/edge)
__global__ void scatter_kernel(const float* __restrict__ hin, float* __restrict__ z,
                               const long long* __restrict__ ei) {
  int tid = blockIdx.x * blockDim.x + threadIdx.x;
  int e = tid >> 5;
  int lane = tid & 31;
  if (e >= N_EDGES) return;
  long long src = ei[e];
  long long dst = ei[N_EDGES + e];
  float4 v = ((const float4*)(hin + src * DIM))[lane];
  float* zd = z + dst * DIM + (lane << 2);
  atomicAdd(zd + 0, v.x);
  atomicAdd(zd + 1, v.y);
  atomicAdd(zd + 2, v.z);
  atomicAdd(zd + 3, v.w);
}

// ---------------- fused MLP: zout = relu(zin@W1 + b1)@W2 + b2 via bf16 WMMA ----
__global__ __launch_bounds__(256) void mlp_kernel(
    const float* __restrict__ zin, float* __restrict__ zout,
    const unsigned short* __restrict__ Wt1, const unsigned short* __restrict__ Wt2,
    const float* __restrict__ b1, const float* __restrict__ b2, int layer) {
  __shared__ __align__(16) unsigned short As[BROWS * LDS_STRIDE];  // act tile  [m][k] bf16
  __shared__ __align__(16) unsigned short Hs[BROWS * LDS_STRIDE];  // hidden    [m][k] bf16

  const int tid  = threadIdx.x;
  const int wave = tid >> 5;
  const int lane = tid & 31;
  const int row0 = blockIdx.x * BROWS;

  // stage A tile, two bf16 per 32-bit LDS store
  for (int i = tid; i < BROWS * (DIM / 2); i += 256) {   // 4096/2 = 2048
    int r = i >> 6, cp = i & 63;                         // cp = column pair
    int gr = row0 + r;
    float2 v = make_float2(0.f, 0.f);
    if (gr < N_NODES) v = ((const float2*)(zin + (size_t)gr * DIM))[cp];
    unsigned p = ((unsigned)f2bf(v.y) << 16) | (unsigned)f2bf(v.x);
    ((unsigned*)As)[r * (LDS_STRIDE / 2) + cp] = p;
  }
  __syncthreads();

  const int stripe = wave >> 2;        // 16-row stripe (0..1)
  const int n0     = (wave & 3) * 32;  // column base: two 16-wide N tiles
  const int lhalf  = lane >> 4;
  const int l15    = lane & 15;
  const unsigned short* arow = As + (stripe * 16 + l15) * LDS_STRIDE;
  const unsigned short* hrow = Hs + (stripe * 16 + l15) * LDS_STRIDE;

  const unsigned short* w1l = Wt1 + (layer << 14);
  const unsigned short* w2l = Wt2 + (layer << 14);
  const unsigned short* w1r0 = w1l + (n0 + l15) * DIM;
  const unsigned short* w1r1 = w1l + (n0 + 16 + l15) * DIM;
  const unsigned short* w2r0 = w2l + (n0 + l15) * DIM;
  const unsigned short* w2r1 = w2l + (n0 + 16 + l15) * DIM;

  // ---- GEMM1: bias preloaded into accumulators (C: N=lane&15, M=r+8*lhalf)
  v8f c0, c1;
  {
    float bv0 = b1[layer * DIM + n0 + l15];
    float bv1 = b1[layer * DIM + n0 + 16 + l15];
    #pragma unroll
    for (int r = 0; r < 8; ++r) { c0[r] = bv0; c1[r] = bv1; }
  }
  #pragma unroll
  for (int ks = 0; ks < 4; ++ks) {
    const int kbase = ks * 32;
    v16bf a  = load_a_frag(arow, kbase, lhalf);
    v16bf bA = load_b_frag(w1r0, kbase, lhalf);
    v16bf bB = load_b_frag(w1r1, kbase, lhalf);
    c0 = __builtin_amdgcn_wmma_f32_16x16x32_bf16(false, a, false, bA, (short)0, c0, false, false);
    c1 = __builtin_amdgcn_wmma_f32_16x16x32_bf16(false, a, false, bB, (short)0, c1, false, false);
  }

  // relu -> Hs (bf16); Hs[m][n] is exactly GEMM2's [M][K]
  #pragma unroll
  for (int r = 0; r < 8; ++r) {
    int m = stripe * 16 + r + (lhalf << 3);
    float v0 = c0[r] > 0.f ? c0[r] : 0.f;
    float v1 = c1[r] > 0.f ? c1[r] : 0.f;
    Hs[m * LDS_STRIDE + n0 + l15]      = f2bf(v0);
    Hs[m * LDS_STRIDE + n0 + 16 + l15] = f2bf(v1);
  }
  __syncthreads();

  // ---- GEMM2
  v8f d0, d1;
  {
    float bv0 = b2[layer * DIM + n0 + l15];
    float bv1 = b2[layer * DIM + n0 + 16 + l15];
    #pragma unroll
    for (int r = 0; r < 8; ++r) { d0[r] = bv0; d1[r] = bv1; }
  }
  #pragma unroll
  for (int ks = 0; ks < 4; ++ks) {
    const int kbase = ks * 32;
    v16bf a  = load_a_frag(hrow, kbase, lhalf);
    v16bf bA = load_b_frag(w2r0, kbase, lhalf);
    v16bf bB = load_b_frag(w2r1, kbase, lhalf);
    d0 = __builtin_amdgcn_wmma_f32_16x16x32_bf16(false, a, false, bA, (short)0, d0, false, false);
    d1 = __builtin_amdgcn_wmma_f32_16x16x32_bf16(false, a, false, bB, (short)0, d1, false, false);
  }

  #pragma unroll
  for (int r = 0; r < 8; ++r) {
    int m = stripe * 16 + r + (lhalf << 3);
    int gr = row0 + m;
    if (gr < N_NODES) {
      zout[(size_t)gr * DIM + n0 + l15]      = d0[r];
      zout[(size_t)gr * DIM + n0 + 16 + l15] = d1[r];
    }
  }
}

// ------------------------------------- per-column mean/var (biased), 4 cols/block
__global__ void bn_stats_kernel(const float* __restrict__ z, float* __restrict__ mean,
                                float* __restrict__ var) {
  __shared__ float red[256][8];
  const int c4 = blockIdx.x;           // 0..31 (4 columns each)
  float s0 = 0.f, s1 = 0.f, s2 = 0.f, s3 = 0.f;
  float q0 = 0.f, q1 = 0.f, q2 = 0.f, q3 = 0.f;
  for (int r = threadIdx.x; r < N_NODES; r += 256) {
    float4 v = ((const float4*)z)[r * (DIM / 4) + c4];
    s0 += v.x; s1 += v.y; s2 += v.z; s3 += v.w;
    q0 += v.x * v.x; q1 += v.y * v.y; q2 += v.z * v.z; q3 += v.w * v.w;
  }
  red[threadIdx.x][0] = s0; red[threadIdx.x][1] = s1;
  red[threadIdx.x][2] = s2; red[threadIdx.x][3] = s3;
  red[threadIdx.x][4] = q0; red[threadIdx.x][5] = q1;
  red[threadIdx.x][6] = q2; red[threadIdx.x][7] = q3;
  __syncthreads();
  for (int off = 128; off > 0; off >>= 1) {
    if (threadIdx.x < off) {
      #pragma unroll
      for (int j = 0; j < 8; ++j) red[threadIdx.x][j] += red[threadIdx.x + off][j];
    }
    __syncthreads();
  }
  if (threadIdx.x < 4) {
    const float invn = 1.0f / (float)N_NODES;
    float m = red[0][threadIdx.x] * invn;
    mean[c4 * 4 + threadIdx.x] = m;
    var[c4 * 4 + threadIdx.x]  = red[0][threadIdx.x + 4] * invn - m * m;
  }
}

// --------------------------------------------------- h = BN(z) (+ relu if not last)
__global__ void bn_apply_kernel(const float* __restrict__ z, float* __restrict__ h,
                                const float* __restrict__ mean, const float* __restrict__ var,
                                const float* __restrict__ gamma, const float* __restrict__ beta,
                                int layer, int do_relu) {
  int i = blockIdx.x * blockDim.x + threadIdx.x;   // over N*D/4
  if (i >= N_NODES * DIM / 4) return;
  int c4 = i & (DIM / 4 - 1);
  float4 zv = ((const float4*)z)[i];
  float4 mv = ((const float4*)mean)[c4];
  float4 vv = ((const float4*)var)[c4];
  float4 gv = ((const float4*)(gamma + layer * DIM))[c4];
  float4 bv = ((const float4*)(beta + layer * DIM))[c4];
  float4 o;
  o.x = (zv.x - mv.x) * rsqrtf(vv.x + BN_EPS_F) * gv.x + bv.x;
  o.y = (zv.y - mv.y) * rsqrtf(vv.y + BN_EPS_F) * gv.y + bv.y;
  o.z = (zv.z - mv.z) * rsqrtf(vv.z + BN_EPS_F) * gv.z + bv.z;
  o.w = (zv.w - mv.w) * rsqrtf(vv.w + BN_EPS_F) * gv.w + bv.w;
  if (do_relu) {
    o.x = fmaxf(o.x, 0.f); o.y = fmaxf(o.y, 0.f);
    o.z = fmaxf(o.z, 0.f); o.w = fmaxf(o.w, 0.f);
  }
  ((float4*)h)[i] = o;
}

// ------------------------------------------------------------------ pooling
__global__ void pool_zero_kernel(float* __restrict__ sums, float* __restrict__ counts) {
  int i = blockIdx.x * blockDim.x + threadIdx.x;
  if (i < N_GRAPHS * DIM) sums[i] = 0.f;
  if (i < N_GRAPHS) counts[i] = 0.f;
}

__global__ void pool_accum_kernel(const float* __restrict__ h, const long long* __restrict__ batch,
                                  float* __restrict__ sums, float* __restrict__ counts) {
  int tid = blockIdx.x * blockDim.x + threadIdx.x;
  int node = tid >> 5, lane = tid & 31;
  if (node >= N_NODES) return;
  long long g = batch[node];
  float4 v = ((const float4*)(h + (size_t)node * DIM))[lane];
  float* sd = sums + g * DIM + (lane << 2);
  atomicAdd(sd + 0, v.x);
  atomicAdd(sd + 1, v.y);
  atomicAdd(sd + 2, v.z);
  atomicAdd(sd + 3, v.w);
  if (lane == 0) atomicAdd(counts + g, 1.0f);
}

__global__ void pool_div_kernel(float* __restrict__ out, const float* __restrict__ sums,
                                const float* __restrict__ counts) {
  int i = blockIdx.x * blockDim.x + threadIdx.x;
  if (i >= N_GRAPHS * DIM) return;
  float c = counts[i >> 7];
  out[i] = sums[i] / fmaxf(c, 1.0f);
}

// ------------------------------------------------------------------ launcher
extern "C" void kernel_launch(void* const* d_in, const int* in_sizes, int n_in,
                              void* d_out, int out_size, void* d_ws, size_t ws_size,
                              hipStream_t stream) {
  const float*      x     = (const float*)d_in[0];
  const long long*  ei    = (const long long*)d_in[1];   // int64 (2, E)
  const long long*  batch = (const long long*)d_in[2];   // int64 (N,)
  const float*      W1    = (const float*)d_in[3];
  const float*      b1    = (const float*)d_in[4];
  const float*      W2    = (const float*)d_in[5];
  const float*      b2    = (const float*)d_in[6];
  const float*      eps   = (const float*)d_in[7];
  const float*      gamma = (const float*)d_in[8];
  const float*      beta  = (const float*)d_in[9];
  float*            out   = (float*)d_out;

  char* ws = (char*)d_ws;
  const size_t FEAT_BYTES = (size_t)N_NODES * DIM * sizeof(float);   // 25.6 MB
  float* h      = (float*)(ws);
  float* z      = (float*)(ws + FEAT_BYTES);
  float* sums   = (float*)(ws + 2 * FEAT_BYTES);
  float* counts = sums + N_GRAPHS * DIM;
  float* mean   = counts + N_GRAPHS;
  float* var    = mean + DIM;
  unsigned short* wt1 = (unsigned short*)(var + DIM);                // 3*128*128 bf16
  unsigned short* wt2 = wt1 + NLAYERS * DIM * DIM;

  const int wconv_blocks   = (NLAYERS * DIM * DIM + 255) / 256;      // 192
  const int zinit_blocks   = (N_NODES * DIM / 4 + 255) / 256;        // 6250
  const int scatter_blocks = (N_EDGES * 32 + 255) / 256;             // 100000
  const int mlp_blocks     = (N_NODES + BROWS - 1) / BROWS;          // 1563
  const int poolz_blocks   = (N_GRAPHS * DIM + 255) / 256;           // 256
  const int poola_blocks   = (N_NODES * 32 + 255) / 256;             // 6250

  wconv_kernel<<<wconv_blocks, 256, 0, stream>>>(W1, W2, wt1, wt2);

  for (int l = 0; l < NLAYERS; ++l) {
    const float* hin = (l == 0) ? x : h;
    zinit_kernel<<<zinit_blocks, 256, 0, stream>>>(hin, z, eps, l);
    scatter_kernel<<<scatter_blocks, 256, 0, stream>>>(hin, z, ei);
    mlp_kernel<<<mlp_blocks, 256, 0, stream>>>(z, z, wt1, wt2, b1, b2, l);
    bn_stats_kernel<<<DIM / 4, 256, 0, stream>>>(z, mean, var);
    bn_apply_kernel<<<zinit_blocks, 256, 0, stream>>>(z, h, mean, var, gamma, beta, l,
                                                      (l < NLAYERS - 1) ? 1 : 0);
  }
  pool_zero_kernel<<<poolz_blocks, 256, 0, stream>>>(sums, counts);
  pool_accum_kernel<<<poola_blocks, 256, 0, stream>>>(h, batch, sums, counts);
  pool_div_kernel<<<poolz_blocks, 256, 0, stream>>>(out, sums, counts);
}
